// AutoregressiveAttention_57964878627115
// MI455X (gfx1250) — matrix operand order repaired and checked by reading, not compiled, not run on hardware
//
#include <hip/hip_runtime.h>
#include <hip/hip_bf16.h>

// ---------------------------------------------------------------------------
// Causal attention, d_model = 1024, B = 4, S = 2048, fp32 in/out.
// Pipeline: (1) QKV projection GEMMs (bf16 WMMA, f32 accum)
//           (2) scores = Q K^T / 32 (lower-triangular tiles only)
//           (3) row softmax with causal mask, P written bf16 in-place
//           (4) O = P V (reduction truncated at the diagonal)
// All matrix math uses v_wmma_f32_16x16x32_bf16 (wave32).
// Per-wave GEMM tile: 64x64 (16 accumulator tiles) => 16 B/lane of operand
// traffic per WMMA, ~180 VGPRs/wave.
// ---------------------------------------------------------------------------

#define DM   1024
#define SEQ  2048
#define NB   4
#define NEGV (-1e9f)

typedef __attribute__((ext_vector_type(16))) __bf16 v16bf;
typedef __attribute__((ext_vector_type(8)))  float  v8f;

union Frag { v16bf v; uint4 q[2]; };

// A operand: 16x32 (MxK) tile from row-major bf16 matrix, leading dim ld.
// lanes 0-15: row M=lane, K = k0..k0+15
// lanes 16-31: row M=lane-16, K = k0+16..k0+31
__device__ __forceinline__ v16bf load_a_bf16(const __bf16* base, int ld,
                                             int row0, int k0, int lane) {
    Frag f;
    int r  = row0 + (lane & 15);
    int kb = k0 + ((lane >> 4) << 4);
    const __bf16* p = base + (size_t)r * ld + kb;
    f.q[0] = *reinterpret_cast<const uint4*>(p);
    f.q[1] = *reinterpret_cast<const uint4*>(p + 8);
    return f.v;
}

// Same A layout but converting from fp32 memory.
__device__ __forceinline__ v16bf load_a_f32(const float* base, int ld,
                                            int row0, int k0, int lane) {
    int r  = row0 + (lane & 15);
    int kb = k0 + ((lane >> 4) << 4);
    const float* p = base + (size_t)r * ld + kb;
    v16bf a;
#pragma unroll
    for (int i = 0; i < 8; ++i) {
        a[i]     = (__bf16)p[i];
        a[8 + i] = (__bf16)p[8 + i];
    }
    return a;
}

// B operand: 32x16 (KxN) tile supplied as Bt row-major (Bt[n][k]), leading dim ld.
// lane holds column N = lane&15, contiguous K = k0 + (lane>>4)*16 .. +15.
__device__ __forceinline__ v16bf load_b_bf16(const __bf16* bt, int ld,
                                             int n0, int k0, int lane) {
    Frag f;
    int n  = n0 + (lane & 15);
    int kb = k0 + ((lane >> 4) << 4);
    const __bf16* p = bt + (size_t)n * ld + kb;
    f.q[0] = *reinterpret_cast<const uint4*>(p);
    f.q[1] = *reinterpret_cast<const uint4*>(p + 8);
    return f.v;
}

__device__ __forceinline__ v16bf load_b_f32(const float* bt, int ld,
                                            int n0, int k0, int lane) {
    int n  = n0 + (lane & 15);
    int kb = k0 + ((lane >> 4) << 4);
    const float* p = bt + (size_t)n * ld + kb;
    v16bf b;
#pragma unroll
    for (int i = 0; i < 16; ++i) b[i] = (__bf16)p[i];
    return b;
}

// ---------------------------------------------------------------------------
// Kernel 1: Q/K/V = X @ W^T + b.  X fp32 [8192,1024], W fp32 [1024,1024]
// row-major (torch Linear => W rows are the reduction vectors: Bt layout).
// Q,K stored row-major bf16 [token][d]; V stored transposed bf16 [b][d][s].
// Wave tile 64(m) x 64(n); block = 4 waves = 256 m x 64 n.
// ---------------------------------------------------------------------------
__global__ __launch_bounds__(128) void qkv_proj_kernel(
    const float* __restrict__ X,
    const float* __restrict__ Wq, const float* __restrict__ bq,
    const float* __restrict__ Wk, const float* __restrict__ bk,
    const float* __restrict__ Wv, const float* __restrict__ bv,
    __bf16* __restrict__ Qb, __bf16* __restrict__ Kb, __bf16* __restrict__ Vt) {
    const int lane  = threadIdx.x & 31;
    const int wave  = threadIdx.x >> 5;
    const int which = blockIdx.z;                       // 0=Q 1=K 2=V
    const int n0 = blockIdx.x * 64;                     // 16 n-groups
    const int m0 = (blockIdx.y * 4 + wave) * 64;        // 128 m-strips of 64

    const float* W    = (which == 0) ? Wq : (which == 1) ? Wk : Wv;
    const float* bias = (which == 0) ? bq : (which == 1) ? bk : bv;

    v8f acc[16] = {};
    for (int k = 0; k < DM; k += 32) {
        v16bf a[4];
#pragma unroll
        for (int i = 0; i < 4; ++i)
            a[i] = load_a_f32(X, DM, m0 + i * 16, k, lane);
#pragma unroll
        for (int j = 0; j < 4; ++j) {
            v16bf bb = load_b_f32(W, DM, n0 + j * 16, k, lane);
#pragma unroll
            for (int i = 0; i < 4; ++i)
                acc[i * 4 + j] = __builtin_amdgcn_wmma_f32_16x16x32_bf16(
                    false, a[i], false, bb, (short)0, acc[i * 4 + j], false, false);
        }
    }

    const int lr = lane & 15;
    const int lh = lane >> 4;
#pragma unroll
    for (int j = 0; j < 4; ++j) {
        const int n = n0 + j * 16 + lr;
        const float bias_n = bias[n];
#pragma unroll
        for (int i = 0; i < 4; ++i) {
            const int mb = m0 + i * 16 + (lh << 3);
            if (which == 0 || which == 1) {
                __bf16* O = (which == 0) ? Qb : Kb;
#pragma unroll
                for (int g = 0; g < 8; ++g)
                    O[(size_t)(mb + g) * DM + n] = (__bf16)(acc[i * 4 + j][g] + bias_n);
            } else {
#pragma unroll
                for (int g = 0; g < 8; ++g) {
                    const int tok = mb + g;
                    const int bb  = tok >> 11;          // /2048
                    const int s   = tok & (SEQ - 1);
                    Vt[((size_t)bb * DM + n) * SEQ + s] = (__bf16)(acc[i * 4 + j][g] + bias_n);
                }
            }
        }
    }
}

// ---------------------------------------------------------------------------
// Kernel 2: scores[b][q][k] = (Q[b,q,:] . K[b,k,:]) / 32, f32.
// Wave tile 64q x 64k; block = 4 waves = 256 q x 64 k. Waves whose whole
// 64x64 tile lies strictly above the diagonal return early; softmax masks
// col>row elementwise, so skipped regions may hold garbage.
// ---------------------------------------------------------------------------
__global__ __launch_bounds__(128) void scores_kernel(
    const __bf16* __restrict__ Qb, const __bf16* __restrict__ Kb,
    float* __restrict__ Sc) {
    const int lane = threadIdx.x & 31;
    const int wave = threadIdx.x >> 5;
    const int b   = blockIdx.z;
    const int qg  = blockIdx.y;                 // 8 groups of 256 queries
    const int kg  = blockIdx.x;                 // 32 groups of 64 keys
    const int kt0 = kg * 64;
    const int q0  = qg * 256 + wave * 64;
    if (kt0 > q0 + 63) return;                  // tile entirely above diagonal

    const __bf16* Qbase = Qb + (size_t)b * SEQ * DM;
    const __bf16* Kbase = Kb + (size_t)b * SEQ * DM;

    v8f acc[16] = {};
    for (int k = 0; k < DM; k += 32) {
        v16bf a[4];
#pragma unroll
        for (int i = 0; i < 4; ++i)
            a[i] = load_a_bf16(Qbase, DM, q0 + i * 16, k, lane);
#pragma unroll
        for (int j = 0; j < 4; ++j) {
            v16bf bb = load_b_bf16(Kbase, DM, kt0 + j * 16, k, lane);
#pragma unroll
            for (int i = 0; i < 4; ++i)
                acc[i * 4 + j] = __builtin_amdgcn_wmma_f32_16x16x32_bf16(
                    false, a[i], false, bb, (short)0, acc[i * 4 + j], false, false);
        }
    }

    const int lr = lane & 15;
    const int lh = lane >> 4;
    float* out = Sc + (size_t)b * SEQ * SEQ;
#pragma unroll
    for (int i = 0; i < 4; ++i) {
        const int mb = q0 + i * 16 + (lh << 3);
#pragma unroll
        for (int j = 0; j < 4; ++j) {
            const int n = kt0 + j * 16 + lr;
#pragma unroll
            for (int g = 0; g < 8; ++g)
                out[(size_t)(mb + g) * SEQ + n] = acc[i * 4 + j][g] * 0.03125f; // 1/sqrt(1024)
        }
    }
}

// ---------------------------------------------------------------------------
// Kernel 3: causal softmax per row. One 256-thread block per (b,q) row.
// Reads 2048 f32 scores; writes 2048 bf16 probabilities IN PLACE over the
// first half of the row (bf16 row stride = 4096 elements). All reads happen
// before the first __syncthreads, all writes after the reductions.
// ---------------------------------------------------------------------------
__global__ __launch_bounds__(256) void softmax_kernel(float* __restrict__ Sc) {
    const int row = blockIdx.x;          // 0..8191
    const int q   = row & (SEQ - 1);
    float* sr = Sc + (size_t)row * SEQ;
    __bf16* pr = reinterpret_cast<__bf16*>(sr);
    const int t = threadIdx.x;

    float vals[8];
    float mx = -3.0e38f;
#pragma unroll
    for (int i = 0; i < 8; ++i) {
        const int c = t + i * 256;
        float x = sr[c];
        x = (c > q) ? NEGV : x;
        vals[i] = x;
        mx = fmaxf(mx, x);
    }

    __shared__ float red[256];
    red[t] = mx;
    __syncthreads();
    for (int off = 128; off > 0; off >>= 1) {
        if (t < off) red[t] = fmaxf(red[t], red[t + off]);
        __syncthreads();
    }
    mx = red[0];
    __syncthreads();

    float sum = 0.f;
#pragma unroll
    for (int i = 0; i < 8; ++i) {
        const float e = __expf(vals[i] - mx);
        vals[i] = e;
        sum += e;
    }
    red[t] = sum;
    __syncthreads();
    for (int off = 128; off > 0; off >>= 1) {
        if (t < off) red[t] += red[t + off];
        __syncthreads();
    }
    const float inv = 1.0f / red[0];
    __syncthreads();

#pragma unroll
    for (int i = 0; i < 8; ++i) {
        const int c = t + i * 256;
        pr[c] = (__bf16)((c > q) ? 0.f : vals[i] * inv);
    }
}

// ---------------------------------------------------------------------------
// Kernel 4: O[b][q][n] = sum_k P[b][q][k] * V[b][k][n].
// A = P (bf16, row stride 4096 inside the f32 score buffer), Bt = Vt[b][n][s].
// Wave tile 64q x 64n; block = 4 waves sharing one q-strip => 64 q x 256 n,
// so the reduction bound kend = q0+64 is uniform across the block and P's
// zeros past the diagonal 64-key tile are never touched.
// ---------------------------------------------------------------------------
__global__ __launch_bounds__(128) void out_gemm_kernel(
    const float* __restrict__ Sc, const __bf16* __restrict__ Vt,
    float* __restrict__ Out) {
    const int lane = threadIdx.x & 31;
    const int wave = threadIdx.x >> 5;
    const int b  = blockIdx.z;
    const int qg = blockIdx.y;                   // 32 q-strips of 64
    const int ng = blockIdx.x;                   // 4 n-groups of 256
    const int q0 = qg * 64;
    const int n0 = ng * 256 + wave * 64;

    const __bf16* P = reinterpret_cast<const __bf16*>(Sc + (size_t)b * SEQ * SEQ);
    const int ldP = SEQ * 2;                     // 4096 bf16 per row (in-place)
    const __bf16* V = Vt + (size_t)b * DM * SEQ;

    const int kend = q0 + 64;                    // keys needed: <= q0+63

    v8f acc[16] = {};
    for (int k = 0; k < kend; k += 32) {
        v16bf a[4];
#pragma unroll
        for (int i = 0; i < 4; ++i)
            a[i] = load_a_bf16(P, ldP, q0 + i * 16, k, lane);
#pragma unroll
        for (int j = 0; j < 4; ++j) {
            v16bf bb = load_b_bf16(V, SEQ, n0 + j * 16, k, lane);
#pragma unroll
            for (int i = 0; i < 4; ++i)
                acc[i * 4 + j] = __builtin_amdgcn_wmma_f32_16x16x32_bf16(
                    false, a[i], false, bb, (short)0, acc[i * 4 + j], false, false);
        }
    }

    const int lr = lane & 15;
    const int lh = lane >> 4;
#pragma unroll
    for (int i = 0; i < 4; ++i) {
        const int mb = q0 + i * 16 + (lh << 3);
#pragma unroll
        for (int j = 0; j < 4; ++j) {
            const int n = n0 + j * 16 + lr;
#pragma unroll
            for (int g = 0; g < 8; ++g)
                Out[((size_t)b * SEQ + mb + g) * DM + n] = acc[i * 4 + j][g];
        }
    }
}

// ---------------------------------------------------------------------------
// Workspace layout (112 MB):
//   [0,16MB)    Q bf16 [4*2048][1024]
//   [16,32MB)   K bf16 [4*2048][1024]
//   [32,48MB)   V^T bf16 [4][1024][2048]
//   [48,112MB)  scores f32 [4][2048][2048]  (P bf16 overlaid in-place)
// ---------------------------------------------------------------------------
extern "C" void kernel_launch(void* const* d_in, const int* in_sizes, int n_in,
                              void* d_out, int out_size, void* d_ws, size_t ws_size,
                              hipStream_t stream) {
    const float* X  = (const float*)d_in[0];
    const float* Wq = (const float*)d_in[1];
    const float* bq = (const float*)d_in[2];
    const float* Wk = (const float*)d_in[3];
    const float* bk = (const float*)d_in[4];
    const float* Wv = (const float*)d_in[5];
    const float* bv = (const float*)d_in[6];
    float* Out = (float*)d_out;

    char* ws = (char*)d_ws;
    __bf16* Qb = (__bf16*)(ws);
    __bf16* Kb = (__bf16*)(ws + (size_t)16 * 1024 * 1024);
    __bf16* Vt = (__bf16*)(ws + (size_t)32 * 1024 * 1024);
    float*  Sc = (float*) (ws + (size_t)48 * 1024 * 1024);

    // 1) QKV projections: grid (n-groups=16, m-groups=32, qkv=3), 4 waves/block
    qkv_proj_kernel<<<dim3(16, 32, 3), 128, 0, stream>>>(
        X, Wq, bq, Wk, bk, Wv, bv, Qb, Kb, Vt);

    // 2) scores (lower-triangular tiles): grid (kg=32, qg=8, b=4)
    scores_kernel<<<dim3(32, 8, NB), 128, 0, stream>>>(Qb, Kb, Sc);

    // 3) causal softmax: one block per row
    softmax_kernel<<<dim3(NB * SEQ), 256, 0, stream>>>(Sc);

    // 4) O = P @ V: grid (ng=4, qg=32, b=4)
    out_gemm_kernel<<<dim3(4, 32, NB), 128, 0, stream>>>(Sc, Vt, Out);
}
